// MLPPredictor_3444563771451
// MI455X (gfx1250) — matrix operand order repaired and checked by reading, not compiled, not run on hardware
//
#include <hip/hip_runtime.h>

// ---------------------------------------------------------------------------
// Fused edge-MLP:  score = W2 @ [relu(LN(W1 @ [h[src]|h[dst]] + b1)) | polar] + b2
// GEMM1 (E x 256 x 256) runs on v_wmma_f32_16x16x32_bf16 (bf16 in, f32 acc).
// ---------------------------------------------------------------------------

typedef __bf16 bf16_t;
typedef __attribute__((ext_vector_type(16))) __bf16 v16bf;
typedef __attribute__((ext_vector_type(8)))  __bf16 v8bf;
typedef __attribute__((ext_vector_type(8)))  float  v8f;
typedef __attribute__((ext_vector_type(4)))  float  v4f;

#define DFEAT     128
#define TWO_D     256
#define HDIM      256
#define W2K       262            // H + 6
#define TM        128            // edges per block
#define XS_STRIDE 40             // bf16 elems per staged x-row (padded, 80 B)
#define YS_STRIDE 260            // f32 elems per y-row (padded)
#define XS_BYTES  (TM * XS_STRIDE * 2)        // 10240
#define YS_BYTES  (TM * YS_STRIDE * 4)        // 133120

// ---------------------------------------------------------------------------
// Prep: convert W1 (f32 [256][256], row = out-feature n, col = k) into bf16
// B-fragments for v_wmma_16x16x32 in contiguous k order (same as the A side):
//   fragment f = ntile*8 + kstep  (ntile 0..15, kstep 0..7), 512 bf16 each.
//   lane l holds W1[n0 + (l&15)][k0 + (l>=16)*16 + j], j = 0..15 (contiguous).
// Main-loop B loads become fully coalesced 2x global_load_b128 per lane.
// ---------------------------------------------------------------------------
__global__ void w1_swizzle_kernel(const float* __restrict__ W1,
                                  bf16_t* __restrict__ W1s) {
    const int tid   = blockIdx.x * blockDim.x + threadIdx.x;   // 0..4095
    const int lane  = tid & 31;
    const int frag  = tid >> 5;        // 0..127
    const int ntile = frag >> 3;       // 0..15
    const int kstep = frag & 7;        // 0..7
    const int n     = ntile * 16 + (lane & 15);
    const int kbase = kstep * 32 + ((lane >> 4) << 4);
    const float* sp = W1 + n * TWO_D + kbase;
    bf16_t*      dp = W1s + frag * 512 + lane * 16;
    v4f f0 = *(const v4f*)(sp + 0);
    v4f f1 = *(const v4f*)(sp + 4);
    v4f f2 = *(const v4f*)(sp + 8);
    v4f f3 = *(const v4f*)(sp + 12);
    v8f a = __builtin_shufflevector(f0, f1, 0, 1, 2, 3, 4, 5, 6, 7);
    v8f b = __builtin_shufflevector(f2, f3, 0, 1, 2, 3, 4, 5, 6, 7);
    *(v8bf*)(dp + 0) = __builtin_convertvector(a, v8bf);
    *(v8bf*)(dp + 8) = __builtin_convertvector(b, v8bf);
}

// ---------------------------------------------------------------------------
// Main fused kernel. 256 threads = 8 wave32s; block tile = 128 edges x 256 H.
// Wave grid 2 (M) x 4 (N): each wave owns 64 edge-rows x 64 out-cols
// = 4x4 WMMA accumulator tiles (128 VGPRs of f32 acc per lane).
// ---------------------------------------------------------------------------
__global__ __launch_bounds__(256) void edge_mlp_kernel(
    const float* __restrict__ h,
    const int*   __restrict__ src,
    const int*   __restrict__ dst,
    const float* __restrict__ polar,
    const bf16_t* __restrict__ W1s,
    const float* __restrict__ b1,
    const float* __restrict__ gamma,
    const float* __restrict__ beta,
    const float* __restrict__ W2,
    const float* __restrict__ b2,
    float* __restrict__ out,
    int E)
{
    extern __shared__ char smem[];
    bf16_t* xs = (bf16_t*)smem;                 // [TM][XS_STRIDE] staged x tile
    float*  ys = (float*)(smem + XS_BYTES);     // [TM][YS_STRIDE] pre-LN tile

    const int tid    = threadIdx.x;
    const int lane   = tid & 31;
    const int wid    = tid >> 5;       // 0..7
    const int wave_m = wid >> 2;       // 0..1  -> rows  [wave_m*64, +64)
    const int wave_n = wid & 3;        // 0..3  -> cols  [wave_n*64, +64)
    const int e_base = blockIdx.x * TM;

    v8f acc[4][4];
#pragma unroll
    for (int mi = 0; mi < 4; ++mi)
#pragma unroll
        for (int ni = 0; ni < 4; ++ni)
            acc[mi][ni] = (v8f)0.f;

    // Staging role: thread -> (edge row, 16-float half of a 32-wide K slab)
    const int er   = tid >> 1;
    const int half = tid & 1;
    int eg = e_base + er;
    if (eg >= E) eg = 0;                         // clamp (results discarded)
    const long srow = (long)src[eg] * DFEAT;
    const long drow = (long)dst[eg] * DFEAT;

    // A-fragment addressing (contiguous k order, same as W1s):
    //   lanes 0-15 : row M=lane,     K = {0..15}
    //   lanes 16-31: row M=lane-16,  K = {16..31}
    const int arow0 = wave_m * 64 + (lane & 15);
    const int asel  = (lane >> 4) * 16;          // bf16-element offset 0 or 16

#pragma unroll 1
    for (int kstep = 0; kstep < 8; ++kstep) {
        // ---- stage x-tile [TM x 32] (gather h rows, f32 -> bf16) ----
        {
            const long rowbase = (kstep < 4) ? srow : drow;
            const float* gp = h + rowbase + (kstep & 3) * 32 + half * 16;
            v4f f0 = *(const v4f*)(gp + 0);
            v4f f1 = *(const v4f*)(gp + 4);
            v4f f2 = *(const v4f*)(gp + 8);
            v4f f3 = *(const v4f*)(gp + 12);
            v8f a = __builtin_shufflevector(f0, f1, 0, 1, 2, 3, 4, 5, 6, 7);
            v8f b = __builtin_shufflevector(f2, f3, 0, 1, 2, 3, 4, 5, 6, 7);
            bf16_t* sp = xs + er * XS_STRIDE + half * 16;
            *(v8bf*)(sp + 0) = __builtin_convertvector(a, v8bf);
            *(v8bf*)(sp + 8) = __builtin_convertvector(b, v8bf);
        }
        __syncthreads();

        // ---- A fragments from LDS (ds_load_b128 pairs, padded stride) ----
        v16bf afrag[4];
#pragma unroll
        for (int mi = 0; mi < 4; ++mi) {
            const bf16_t* ap = xs + (arow0 + mi * 16) * XS_STRIDE + asel;
            v8bf lo = *(const v8bf*)(ap +  0);   // K 0..7   (or 16..23)
            v8bf hi = *(const v8bf*)(ap +  8);   // K 8..15  (or 24..31)
            afrag[mi] = __builtin_shufflevector(lo, hi,
                0, 1, 2, 3, 4, 5, 6, 7, 8, 9, 10, 11, 12, 13, 14, 15);
        }

        // ---- B fragments (pre-swizzled bf16, coalesced) + 16 WMMAs ----
#pragma unroll
        for (int ni = 0; ni < 4; ++ni) {
            const int ntile = wave_n * 4 + ni;                  // 0..15
            const bf16_t* bp = W1s + (ntile * 8 + kstep) * 512 + lane * 16;
            v8bf b0 = *(const v8bf*)(bp + 0);
            v8bf b1v = *(const v8bf*)(bp + 8);
            v16bf bfrag = __builtin_shufflevector(b0, b1v,
                0, 1, 2, 3, 4, 5, 6, 7, 8, 9, 10, 11, 12, 13, 14, 15);
#pragma unroll
            for (int mi = 0; mi < 4; ++mi) {
                acc[mi][ni] = __builtin_amdgcn_wmma_f32_16x16x32_bf16(
                    false, afrag[mi], false, bfrag,
                    (short)0, acc[mi][ni], false, false);
            }
        }
        __syncthreads();   // xs reuse next iteration
    }

    // ---- +b1, scatter accumulators to padded LDS y-tile ----
    {
        const int rowb = wave_m * 64 + ((lane >> 4) << 3);   // +8 for hi lanes
        const int c15  = lane & 15;
#pragma unroll
        for (int ni = 0; ni < 4; ++ni) {
            const int col = wave_n * 64 + ni * 16 + c15;
            const float bb = b1[col];
#pragma unroll
            for (int mi = 0; mi < 4; ++mi) {
#pragma unroll
                for (int r = 0; r < 8; ++r) {
                    ys[(rowb + mi * 16 + r) * YS_STRIDE + col] =
                        acc[mi][ni][r] + bb;
                }
            }
        }
    }
    __syncthreads();

    // ---- LayerNorm + ReLU + W2 (C=2) + polar tail: 1 thread per edge ----
    if (tid < TM) {
        const int row = tid;
        const int e   = e_base + row;
        const float* yr = ys + row * YS_STRIDE;
        float s = 0.f, ss = 0.f;
#pragma unroll 4
        for (int i = 0; i < HDIM; ++i) {
            const float v = yr[i];
            s  += v;
            ss = fmaf(v, v, ss);
        }
        const float mean = s * (1.0f / HDIM);
        const float var  = fmaxf(ss * (1.0f / HDIM) - mean * mean, 0.f);
        const float rstd = __frsqrt_rn(var + 1e-5f);
        float a0 = b2[0], a1 = b2[1];
#pragma unroll 4
        for (int i = 0; i < HDIM; ++i) {
            float yn = fmaf((yr[i] - mean) * rstd, gamma[i], beta[i]);
            yn = fmaxf(yn, 0.f);
            a0 = fmaf(yn, W2[i], a0);
            a1 = fmaf(yn, W2[W2K + i], a1);
        }
        if (e < E) {
#pragma unroll
            for (int j = 0; j < 6; ++j) {
                const float p = polar[(long)e * 6 + j];
                a0 = fmaf(p, W2[HDIM + j], a0);
                a1 = fmaf(p, W2[W2K + HDIM + j], a1);
            }
            out[(long)e * 2 + 0] = a0;
            out[(long)e * 2 + 1] = a1;
        }
    }
}

// ---------------------------------------------------------------------------
extern "C" void kernel_launch(void* const* d_in, const int* in_sizes, int n_in,
                              void* d_out, int out_size, void* d_ws, size_t ws_size,
                              hipStream_t stream) {
    const float* h     = (const float*)d_in[0];
    const int*   src   = (const int*)  d_in[1];
    const int*   dst   = (const int*)  d_in[2];
    const float* polar = (const float*)d_in[3];
    const float* W1    = (const float*)d_in[4];
    const float* b1    = (const float*)d_in[5];
    const float* gamma = (const float*)d_in[6];
    const float* beta  = (const float*)d_in[7];
    const float* W2    = (const float*)d_in[8];
    const float* b2    = (const float*)d_in[9];
    float* out   = (float*)d_out;
    bf16_t* W1s  = (bf16_t*)d_ws;          // 256*256*2 = 128 KB of scratch
    const int E  = in_sizes[1];

    // 1) swizzle W1 into WMMA B-fragment layout (bf16)
    w1_swizzle_kernel<<<16, 256, 0, stream>>>(W1, W1s);

    // 2) fused gather + GEMM1(WMMA) + LN + ReLU + GEMM2
    const int grid = (E + TM - 1) / TM;
    const size_t smem_bytes = (size_t)XS_BYTES + (size_t)YS_BYTES;  // ~140 KB
    edge_mlp_kernel<<<grid, 256, smem_bytes, stream>>>(
        h, src, dst, polar, W1s, b1, gamma, beta, W2, b2, out, E);
    (void)n_in; (void)out_size; (void)ws_size;
}
